// Attention_609885356930
// MI455X (gfx1250) — matrix operand order repaired and checked by reading, not compiled, not run on hardware
//
#include <hip/hip_runtime.h>
#include <hip/hip_bf16.h>
#include <stdint.h>

// ---------------------------------------------------------------------------
// Problem: B=32, S=2048, D=1024, SIZE(K-out)=1024
//   term1[b,s,k] = WO[k,:] . inputs[b,s,:]      (big GEMM, 137.4 GFLOP)
//   term2[b,k]   = WG[k,:] . g[b,:]
//   out[b,s]     = sum_k v[k] * tanh(term1 + term2)
// Strategy: bf16 WMMA (f32 accum) GEMM fused with tanh + v-dot epilogue.
//   - inputs tile converted to bf16 once into LDS (reused across all N)
//   - WO pre-converted to bf16, streamed via global_load_async_to_lds_b128
//     double buffering (ASYNCcnt pipeline)
//   - fragment loads software-pipelined one K-step ahead of the WMMAs
// ---------------------------------------------------------------------------

typedef __attribute__((ext_vector_type(16))) __bf16 v16bf;
typedef __attribute__((ext_vector_type(8)))  float  v8f;
typedef __attribute__((ext_vector_type(4)))  int    v4i_;   // b128 payload type

union FragAB { v16bf v; uint4 q[2]; };

#if defined(__HIP_DEVICE_COMPILE__) && \
    __has_builtin(__builtin_amdgcn_global_load_async_to_lds_b128) && \
    __has_builtin(__builtin_amdgcn_s_wait_asynccnt)
#define HAVE_ASYNC 1
#else
#define HAVE_ASYNC 0
#endif

__device__ __forceinline__ unsigned short f2bf(float f) {
  union { float f; unsigned u; } x; x.f = f;
  unsigned r = x.u + 0x7FFFu + ((x.u >> 16) & 1u);   // round-to-nearest-even
  return (unsigned short)(r >> 16);
}

// --- prep: term2[b,k] = WG[k,:] . g[b,:] -----------------------------------
__global__ void term2_kernel(const float* __restrict__ WG,
                             const float* __restrict__ g,
                             float* __restrict__ term2) {
  const int k = blockIdx.x * 256 + threadIdx.x;   // 0..1023
  const int b = blockIdx.y;                        // 0..31
  const float4* wg = (const float4*)(WG + (size_t)k * 1024);
  const float4* gb = (const float4*)(g  + (size_t)b * 1024);
  float s = 0.f;
  for (int i = 0; i < 256; ++i) {
    float4 a = wg[i], c = gb[i];
    s += a.x * c.x + a.y * c.y + a.z * c.z + a.w * c.w;
  }
  term2[b * 1024 + k] = s;
}

// --- prep: WO f32 -> bf16 (once; hot loop streams bf16 from L2) ------------
__global__ void cvt_kernel(const float* __restrict__ WO,
                           unsigned short* __restrict__ WOb) {
  const int i = (blockIdx.x * 256 + threadIdx.x) * 4;
  float4 f = *(const float4*)(WO + i);
  unsigned lo = (unsigned)f2bf(f.x) | ((unsigned)f2bf(f.y) << 16);
  unsigned hi = (unsigned)f2bf(f.z) | ((unsigned)f2bf(f.w) << 16);
  *(uint2*)(WOb + i) = make_uint2(lo, hi);
}

// --- main fused kernel -----------------------------------------------------
// Block tile: M=64 rows (one b, 64 consecutive s), N=1024 in 128-chunks,
// K(D)=1024 in 128-wide double-buffered LDS WO chunks. 8 waves = 4(M) x 2(N).
#define A_STRIDE 1032          // 1024 + 8 bf16 pad (+4 dwords -> conflict-free)
#define B_STRIDE 136           // 128 + 8 bf16 pad
#define B_CHUNK  (128 * B_STRIDE)

// Issue one 128x128 bf16 WO chunk copy into an LDS buffer (32 KB).
__device__ __forceinline__ void stage_issue(const unsigned short* __restrict__ WOb,
                                            unsigned short* bufLds,
                                            int n0, int k0, int tid) {
  const unsigned short* src = WOb + (size_t)n0 * 1024 + k0;
#pragma unroll
  for (int j = 0; j < 8; ++j) {
    int flat = j * 256 + tid;               // uint4 (8xbf16) units, 2048 total
    int row = flat >> 4, cq = flat & 15;
#if HAVE_ASYNC
    __builtin_amdgcn_global_load_async_to_lds_b128(
        (__attribute__((address_space(1))) v4i_*)(uintptr_t)(src + (size_t)row * 1024 + cq * 8),
        (__attribute__((address_space(3))) v4i_*)(uintptr_t)(bufLds + row * B_STRIDE + cq * 8),
        0, 0);
#else
    uint4 d = *(const uint4*)(src + (size_t)row * 1024 + cq * 8);
    *(uint4*)(bufLds + row * B_STRIDE + cq * 8) = d;
#endif
  }
}

__device__ __forceinline__ void stage_wait() {
#if HAVE_ASYNC
  __builtin_amdgcn_s_wait_asynccnt(0);
#endif
}

// Load A fragment (16x32) + this wave's 4 B fragments (32x16 each) from LDS,
// addressed exactly per the CDNA5 16-bit WMMA VGPR layouts.
__device__ __forceinline__ void load_frags(FragAB& a, FragAB (&bf)[4],
                                           const unsigned short* Arow,
                                           const unsigned short* Bcur,
                                           int kabs, int klocal,
                                           int wn, int r, int h) {
  const unsigned short* ap = Arow + kabs + h * 8;
  a.q[0] = *(const uint4*)(ap);        // K = h*8 + 0..7    (VGPR 0-3)
  a.q[1] = *(const uint4*)(ap + 16);   // K = 16+h*8 + 0..7 (VGPR 4-7)
#pragma unroll
  for (int t = 0; t < 4; ++t) {
    const unsigned short* bp =
        Bcur + (wn * 64 + t * 16 + r) * B_STRIDE + klocal + h * 8;
    bf[t].q[0] = *(const uint4*)(bp);  // K = h*8 + 0..7
    bf[t].q[1] = *(const uint4*)(bp + 16);
  }
}

__global__ void __launch_bounds__(256)
fused_kernel(const float* __restrict__ inputs,
             const unsigned short* __restrict__ WOb,
             const float* __restrict__ term2,
             const float* __restrict__ vvec,
             float* __restrict__ out) {
  extern __shared__ char smem[];
  unsigned short* As  = (unsigned short*)smem;        // 64 x A_STRIDE
  unsigned short* Bs  = As + 64 * A_STRIDE;           // 2 x 128 x B_STRIDE
  float*          t2s = (float*)(Bs + 2 * B_CHUNK);
  float*          vs  = t2s + 1024;

  const int tid  = threadIdx.x;
  const int lane = tid & 31;
  const int wave = tid >> 5;      // 0..7
  const int wm   = wave & 3;      // M group (16 rows each)
  const int wn   = wave >> 2;     // N half (64 cols each)
  const int r    = lane & 15;
  const int h    = lane >> 4;

  const int b  = blockIdx.y;
  const int s0 = blockIdx.x * 64;

  // stage term2 row + v into LDS
  for (int i = tid; i < 1024; i += 256) {
    t2s[i] = term2[b * 1024 + i];
    vs[i]  = vvec[i];
  }

  // convert this block's 64x1024 f32 input tile -> bf16 LDS (once)
  const float* Ain = inputs + ((size_t)b * 2048 + s0) * 1024;
  for (int j = 0; j < 64; ++j) {
    int flat = (j * 256 + tid) * 4;
    int row = flat >> 10, col = flat & 1023;
    float4 f = *(const float4*)(Ain + (size_t)row * 1024 + col);
    unsigned lo = (unsigned)f2bf(f.x) | ((unsigned)f2bf(f.y) << 16);
    unsigned hi = (unsigned)f2bf(f.z) | ((unsigned)f2bf(f.w) << 16);
    *(uint2*)(As + row * A_STRIDE + col) = make_uint2(lo, hi);
  }

  float rowpart[8];
#pragma unroll
  for (int i = 0; i < 8; ++i) rowpart[i] = 0.f;

  const unsigned short* Arow = As + (wm * 16 + r) * A_STRIDE;
  const v8f vzero = {};

  for (int n0 = 0; n0 < 1024; n0 += 128) {
    v8f acc[4];
#pragma unroll
    for (int t = 0; t < 4; ++t) acc[t] = vzero;

    stage_issue(WOb, Bs, n0, 0, tid);           // prologue: chunk 0 -> buf 0

    for (int k0 = 0; k0 < 1024; k0 += 128) {
      const int cur = (k0 >> 7) & 1;
      const unsigned short* Bcur = Bs + cur * B_CHUNK;

      stage_wait();        // this wave's in-flight copy (chunk 'cur') done
      __syncthreads();     // all waves' copies done; buf cur^1 fully consumed
      if (k0 + 128 < 1024)
        stage_issue(WOb, Bs + (cur ^ 1) * B_CHUNK, n0, k0 + 128, tid);

      // Software-pipelined fragment double buffer: loads for kk+32 are
      // issued before the WMMAs of kk, so the matrix pipe never waits on a
      // just-issued ds_load (partial s_wait_dscnt instead of full).
      FragAB a0, b0[4];
      load_frags(a0, b0, Arow, Bcur, k0, 0, wn, r, h);
#pragma unroll
      for (int kk = 0; kk < 128; kk += 32) {
        FragAB a1, b1[4];
        if (kk < 96)
          load_frags(a1, b1, Arow, Bcur, k0 + kk + 32, kk + 32, wn, r, h);
#pragma unroll
        for (int t = 0; t < 4; ++t)
          acc[t] = __builtin_amdgcn_wmma_f32_16x16x32_bf16(
              false, a0.v, false, b0[t].v, (short)0, acc[t], false, false);
        if (kk < 96) {
          a0 = a1;
#pragma unroll
          for (int t = 0; t < 4; ++t) b0[t] = b1[t];
        }
      }
    }

    // fused epilogue: rowpart[m] += v[n] * tanh(acc + term2[n])
    // C layout: lane r = N%16, VGPR i = M row (lanes 16-31: M+8)
#pragma unroll
    for (int t = 0; t < 4; ++t) {
      int n = n0 + wn * 64 + t * 16 + r;
      float t2 = t2s[n], vv = vs[n];
#pragma unroll
      for (int i = 0; i < 8; ++i)
        rowpart[i] += vv * tanhf(acc[t][i] + t2);
    }
  }

  // reduce the 16 N-lanes sharing the same h, then 2 lanes/wave store 8 rows
#pragma unroll
  for (int m = 1; m < 16; m <<= 1) {
#pragma unroll
    for (int i = 0; i < 8; ++i) rowpart[i] += __shfl_xor(rowpart[i], m, 32);
  }
  if (r == 0) {
    float* orow = out + (size_t)b * 2048 + s0 + wm * 16 + h * 8;
#pragma unroll
    for (int i = 0; i < 8; ++i) orow[i] = rowpart[i];
  }
}

// ---------------------------------------------------------------------------
extern "C" void kernel_launch(void* const* d_in, const int* in_sizes, int n_in,
                              void* d_out, int out_size, void* d_ws, size_t ws_size,
                              hipStream_t stream) {
  const float* inputs = (const float*)d_in[0];   // [32,2048,1024] f32
  const float* g      = (const float*)d_in[1];   // [32,1024]      f32
  const float* WO     = (const float*)d_in[2];   // [1024,1024]    f32
  const float* WG     = (const float*)d_in[3];   // [1024,1024]    f32
  const float* v      = (const float*)d_in[4];   // [1,1024]       f32
  float* out = (float*)d_out;                    // [32,2048]      f32

  float*          term2 = (float*)d_ws;                                   // 128 KB
  unsigned short* WOb   = (unsigned short*)((char*)d_ws + 32 * 1024 * 4); // 2 MB

  term2_kernel<<<dim3(4, 32), 256, 0, stream>>>(WG, g, term2);
  cvt_kernel<<<1024, 256, 0, stream>>>(WO, WOb);

  size_t smem = (size_t)(64 * A_STRIDE + 2 * B_CHUNK) * sizeof(unsigned short)
              + 2 * 1024 * sizeof(float);   // ~210 KB of the 320 KB WGP LDS
  fused_kernel<<<dim3(32, 32), 256, smem, stream>>>(inputs, WOb, term2, v, out);
}
